// context_bridge_40235253629583
// MI455X (gfx1250) — matrix operand order repaired and checked by reading, not compiled, not run on hardware
//
#include <hip/hip_runtime.h>
#include <hip/hip_bf16.h>

#define USE_ASYNC_STAGE 1

typedef _Float16 h8v  __attribute__((ext_vector_type(8)));
typedef _Float16 h16v __attribute__((ext_vector_type(16)));
typedef float    f8v  __attribute__((ext_vector_type(8)));
typedef __attribute__((address_space(3))) char lds_char;

static constexpr int    Bn  = 4;
static constexpr int    Dn  = 512;
static constexpr int    Nn  = 4096;   // 64*64 spatial
static constexpr size_t CTXE = (size_t)Dn * Dn;       // 262144 elements per [D,D]

__device__ __forceinline__ void async_b128(void* lds_dst, const void* gsrc) {
#if USE_ASYNC_STAGE
    asm volatile("global_load_async_to_lds_b128 %0, %1, off"
                 :: "v"((lds_char*)lds_dst), "v"(gsrc) : "memory");
#else
    *(h8v*)lds_dst = *(const h8v*)gsrc;
#endif
}

__device__ __forceinline__ void wait_async() {
#if USE_ASYNC_STAGE
#if __has_builtin(__builtin_amdgcn_s_wait_asynccnt)
    __builtin_amdgcn_s_wait_asynccnt(0);
#else
    asm volatile("s_wait_asynccnt 0" ::: "memory");
#endif
#endif
}

// ---------------------------------------------------------------------------
// WMMA GEMM: C[M,N] = A[M,K] * B[K,N]  (or * B'[K,N] with B'[k,n]=B[n,k] when
// TRANSB). A,B f16 row-major; C f16 or f32 row-major; optional per-row bias.
// Block tile 64(M) x 256(N), K-step 32, double-buffered LDS staging with
// async global->LDS copies overlapped against WMMA of the previous tile.
// 256 threads = 8 waves; each wave owns a 32x64 subtile = 2x4 fragments of
// v_wmma_f32_16x16x32_f16. Requires M%64==0, N%256==0, K%32==0.
// ---------------------------------------------------------------------------
template<bool TRANSB, bool OUTF16>
__global__ __launch_bounds__(256)
void gemm_wmma(const _Float16* __restrict__ A, size_t sA, int lda,
               const _Float16* __restrict__ B, size_t sB, int ldb,
               void* __restrict__ Cout, size_t sC, int ldc,
               const float* __restrict__ bias, int K)
{
    __shared__ __align__(16) _Float16 sAt[2][64][40];    // pad 8 halves; rows 16B-aligned
    __shared__ __align__(16) _Float16 sBt[2][256][40];   // stored [n][k]

    const int tid  = threadIdx.x;
    const int bb   = blockIdx.z;
    const _Float16* Ab = A + (size_t)bb * sA;
    const _Float16* Bb = B + (size_t)bb * sB;
    const int m0 = blockIdx.y * 64;
    const int n0 = blockIdx.x * 256;

    const int wv   = tid >> 5;
    const int lane = tid & 31;
    const int wm   = (wv & 1) * 32;       // wave M offset in tile
    const int wn   = (wv >> 1) * 64;      // wave N offset in tile

    f8v acc[2][4];
    #pragma unroll
    for (int mi = 0; mi < 2; ++mi)
        #pragma unroll
        for (int ni = 0; ni < 4; ++ni)
            #pragma unroll
            for (int r = 0; r < 8; ++r) acc[mi][ni][r] = 0.0f;

    // tile-load work split (both variants' indices precomputed)
    const int arow  = tid >> 2;            // 0..63
    const int ako   = (tid & 3) * 8;       // 0,8,16,24
    const int bkl   = tid >> 3;            // normal: 0..31 (k row)
    const int bnoff = (tid & 7) * 32;      // normal: 0..224

    auto stage = [&](int kk, int p) {
        // A tile 64x32: per-lane 16B async global->LDS
        async_b128(&sAt[p][arow][ako], &Ab[(size_t)(m0 + arow) * lda + kk + ako]);
        if (TRANSB) {
            // row n of B supplies the k-run directly: contiguous 64B per thread
            const _Float16* src = &Bb[(size_t)(n0 + tid) * ldb + kk];
            #pragma unroll
            for (int j = 0; j < 4; ++j)
                async_b128(&sBt[p][tid][j * 8], src + j * 8);
        } else {
            // register transpose into [n][k]
            const _Float16* src = &Bb[(size_t)(kk + bkl) * ldb + n0 + bnoff];
            h8v r[4];
            #pragma unroll
            for (int j = 0; j < 4; ++j) r[j] = *(const h8v*)(src + j * 8);
            #pragma unroll
            for (int j = 0; j < 4; ++j)
                #pragma unroll
                for (int e = 0; e < 8; ++e)
                    sBt[p][bnoff + j * 8 + e][bkl] = r[j][e];
        }
    };

    // prologue: stage tile 0 into buffer 0
    stage(0, 0);
    wait_async();
    __syncthreads();

    int p = 0;
    for (int kk = 0; kk < K; kk += 32, p ^= 1) {
        // stage next tile into the other buffer while computing this one
        if (kk + 32 < K) stage(kk + 32, p ^ 1);
        if (kk + 64 < K) {  // prefetch the tile after next (global_prefetch_b8)
            __builtin_prefetch(&Ab[(size_t)(m0 + arow) * lda + kk + 64 + ako], 0, 1);
            __builtin_prefetch(&Bb[TRANSB ? ((size_t)(n0 + tid) * ldb + kk + 64)
                                          : ((size_t)(kk + 64 + bkl) * ldb + n0 + bnoff)], 0, 1);
        }

        // ---- fragments per documented CDNA5 wave32 layouts
        const int kbA = (lane & 16) ? 8 : 0;    // A: lanes>=16 hold K 8..15 / 24..31
        const int kbB = (lane & 16) ? 8 : 0;    // B: lanes>=16 hold K 8..15 / 24..31
        union { h16v v; h8v h[2]; } af[2], bf[4];
        #pragma unroll
        for (int mi = 0; mi < 2; ++mi) {
            const int row = wm + mi * 16 + (lane & 15);
            af[mi].h[0] = *(const h8v*)&sAt[p][row][kbA];
            af[mi].h[1] = *(const h8v*)&sAt[p][row][kbA + 16];
        }
        #pragma unroll
        for (int ni = 0; ni < 4; ++ni) {
            const int col = wn + ni * 16 + (lane & 15);
            bf[ni].h[0] = *(const h8v*)&sBt[p][col][kbB];
            bf[ni].h[1] = *(const h8v*)&sBt[p][col][kbB + 16];
        }
        #pragma unroll
        for (int mi = 0; mi < 2; ++mi)
            #pragma unroll
            for (int ni = 0; ni < 4; ++ni)
                acc[mi][ni] = __builtin_amdgcn_wmma_f32_16x16x32_f16(
                    false, af[mi].v, false, bf[ni].v,
                    (short)0, acc[mi][ni], false, false);

        // drain our async writes to buf p^1, then one barrier per iteration:
        // guarantees everyone finished reading buf p (DScnt drained at signal)
        // and staged data for buf p^1 is visible before the next compute.
        wait_async();
        __syncthreads();
    }

    // ---- writeback: C/D layout — VGPR r -> M=r (lanes<16) / r+8 (lanes>=16)
    const int rofs = (lane & 16) >> 1;   // 0 or 8
    const int col  = lane & 15;
    #pragma unroll
    for (int mi = 0; mi < 2; ++mi)
        #pragma unroll
        for (int ni = 0; ni < 4; ++ni)
            #pragma unroll
            for (int r = 0; r < 8; ++r) {
                const int m = m0 + wm + mi * 16 + r + rofs;
                const int n = n0 + wn + ni * 16 + col;
                float v = acc[mi][ni][r] + (bias ? bias[m] : 0.0f);
                const size_t off = (size_t)bb * sC + (size_t)m * ldc + n;
                if (OUTF16) ((_Float16*)Cout)[off] = (_Float16)v;
                else        ((float*)Cout)[off]    = v;
            }
}

// ---------------------------------------------------------------------------
// elementwise / reduction helpers
// ---------------------------------------------------------------------------
__global__ void f32_to_f16(const float* __restrict__ in, _Float16* __restrict__ out,
                           size_t n) {
    size_t i = (size_t)blockIdx.x * 256 + threadIdx.x;
    const size_t stride = (size_t)gridDim.x * 256;
    for (; i < n; i += stride) out[i] = (_Float16)in[i];
}

// out[o,c] = sum_d w2[o,d] * w1[d,c]   (w2: [512,512], w1: [512,C])
__global__ __launch_bounds__(256)
void eff_weight(const float* __restrict__ w2, const float* __restrict__ w1,
                _Float16* __restrict__ out, int C) {
    const int idx = blockIdx.x * 256 + threadIdx.x;
    if (idx >= 512 * C) return;
    const int o = idx / C, c = idx % C;
    float s = 0.f;
    for (int d = 0; d < 512; ++d) s += w2[o * 512 + d] * w1[d * C + c];
    out[idx] = (_Float16)s;
}

// out[o] = sum_d w2[o,d]*mapb[d] + b2[o]
__global__ void eff_bias(const float* __restrict__ w2, const float* __restrict__ mb,
                         const float* __restrict__ b2, float* __restrict__ out) {
    const int o = blockIdx.x * 256 + threadIdx.x;
    if (o >= 512) return;
    float s = b2[o];
    for (int d = 0; d < 512; ++d) s += w2[o * 512 + d] * mb[d];
    out[o] = s;
}

// in-place softmax over the 4096-wide row (spatial softmax of K)
__global__ __launch_bounds__(256) void softmax_row(_Float16* __restrict__ d) {
    _Float16* p = d + (size_t)blockIdx.x * Nn;
    __shared__ float red[256];
    const int t = threadIdx.x;
    float mx = -3.0e30f;
    for (int i = t; i < Nn; i += 256) mx = fmaxf(mx, (float)p[i]);
    red[t] = mx; __syncthreads();
    for (int s = 128; s > 0; s >>= 1) { if (t < s) red[t] = fmaxf(red[t], red[t + s]); __syncthreads(); }
    mx = red[0]; __syncthreads();
    float sm = 0.f;
    for (int i = t; i < Nn; i += 256) sm += __expf((float)p[i] - mx);
    red[t] = sm; __syncthreads();
    for (int s = 128; s > 0; s >>= 1) { if (t < s) red[t] += red[t + s]; __syncthreads(); }
    const float inv = 1.0f / red[0];
    for (int i = t; i < Nn; i += 256) p[i] = (_Float16)(__expf((float)p[i] - mx) * inv);
}

// in-place softmax over the 512 channels for each (b, n) column of Q
__global__ void softmax_col(_Float16* __restrict__ d) {
    const int b = blockIdx.y;
    const int n = blockIdx.x * blockDim.x + threadIdx.x;
    _Float16* p = d + (size_t)b * Dn * Nn + n;
    float mx = -3.0e30f;
    for (int i = 0; i < Dn; ++i) mx = fmaxf(mx, (float)p[(size_t)i * Nn]);
    float sm = 0.f;
    for (int i = 0; i < Dn; ++i) sm += __expf((float)p[(size_t)i * Nn] - mx);
    const float inv = 1.0f / sm;
    for (int i = 0; i < Dn; ++i) {
        const size_t o = (size_t)i * Nn;
        p[o] = (_Float16)(__expf((float)p[o] - mx) * inv);
    }
}

// p[b*3+i] = mean(ctx_i[b]);  grid = 12 blocks, block = b*3+i
__global__ __launch_bounds__(256)
void se_pool(const float* __restrict__ ctxAll, float* __restrict__ p) {
    const int blk = blockIdx.x;
    const int bb = blk / 3, ii = blk % 3;
    const float* base = ctxAll + ((size_t)ii * Bn + bb) * CTXE;
    __shared__ float red[256];
    float s = 0.f;
    for (int i = threadIdx.x; i < (int)CTXE; i += 256) s += base[i];
    red[threadIdx.x] = s; __syncthreads();
    for (int st = 128; st > 0; st >>= 1) {
        if (threadIdx.x < st) red[threadIdx.x] += red[threadIdx.x + st];
        __syncthreads();
    }
    if (threadIdx.x == 0) p[blk] = red[0] * (1.0f / (float)CTXE);
}

__global__ void se_gate(const float* __restrict__ p,
                        const float* __restrict__ fc1w, const float* __restrict__ fc1b,
                        const float* __restrict__ fc2w, const float* __restrict__ fc2b,
                        float* __restrict__ g) {
    const int bb = threadIdx.x;
    if (bb >= Bn) return;
    float pv[3], h[3];
    for (int j = 0; j < 3; ++j) pv[j] = p[bb * 3 + j];
    for (int i = 0; i < 3; ++i) {
        float s = fc1b[i];
        for (int j = 0; j < 3; ++j) s += pv[j] * fc1w[i * 3 + j];
        h[i] = fmaxf(s, 0.f);
    }
    for (int i = 0; i < 3; ++i) {
        float s = fc2b[i];
        for (int j = 0; j < 3; ++j) s += h[j] * fc2w[i * 3 + j];
        g[bb * 3 + i] = 1.0f / (1.0f + __expf(-s));
    }
}

// combined ctx (SE conv over 3 gated contexts), written TRANSPOSED in f16:
// ctxT[b][v][k] = sum_i convw[i]*g[b,i]*ctx_i[b][k][v] + convb
__global__ __launch_bounds__(256)
void se_combine(const float* __restrict__ ctxAll, const float* __restrict__ g,
                const float* __restrict__ convw, const float* __restrict__ convb,
                _Float16* __restrict__ ctxT) {
    const size_t idx = (size_t)blockIdx.x * 256 + threadIdx.x;  // < Bn*CTXE
    const int bb = (int)(idx >> 18);
    const int j  = (int)(idx & (CTXE - 1));
    const int k = j >> 9, v = j & 511;
    float s = convb[0];
    #pragma unroll
    for (int i = 0; i < 3; ++i)
        s += convw[i] * g[bb * 3 + i] * ctxAll[((size_t)i * Bn + bb) * CTXE + j];
    ctxT[(size_t)bb * CTXE + (size_t)v * Dn + k] = (_Float16)s;
}

// ---------------------------------------------------------------------------
extern "C" void kernel_launch(void* const* d_in, const int* in_sizes, int n_in,
                              void* d_out, int out_size, void* d_ws, size_t ws_size,
                              hipStream_t stream) {
    // ---- inputs (setup_inputs order)
    const float* x1     = (const float*)d_in[0];
    const float* x2     = (const float*)d_in[1];
    const float* x3     = (const float*)d_in[2];
    const float* map1_w = (const float*)d_in[3];
    const float* map1_b = (const float*)d_in[4];
    const float* map2_w = (const float*)d_in[5];
    const float* map2_b = (const float*)d_in[6];
    const float* kqv_w[3][3] = {  // [branch][k/q/v]
        {(const float*)d_in[7],  (const float*)d_in[9],  (const float*)d_in[11]},
        {(const float*)d_in[13], (const float*)d_in[15], (const float*)d_in[17]},
        {(const float*)d_in[19], (const float*)d_in[21], (const float*)d_in[23]}};
    const float* kqv_b[3][3] = {
        {(const float*)d_in[8],  (const float*)d_in[10], (const float*)d_in[12]},
        {(const float*)d_in[14], (const float*)d_in[16], (const float*)d_in[18]},
        {(const float*)d_in[20], (const float*)d_in[22], (const float*)d_in[24]}};
    const float* se_fc1_w  = (const float*)d_in[25];
    const float* se_fc1_b  = (const float*)d_in[26];
    const float* se_fc2_w  = (const float*)d_in[27];
    const float* se_fc2_b  = (const float*)d_in[28];
    const float* se_conv_w = (const float*)d_in[29];
    const float* se_conv_b = (const float*)d_in[30];
    const float* rev1_w    = (const float*)d_in[31];
    const float* rev1_b    = (const float*)d_in[32];
    const float* rev2_w    = (const float*)d_in[33];
    const float* rev2_b    = (const float*)d_in[34];

    // ---- workspace layout (each region 256-aligned)
    char* W = (char*)d_ws;
    size_t off = 0;
    auto alloc = [&](size_t bytes) { char* p = W + off; off += (bytes + 255) & ~(size_t)255; return p; };
    _Float16* XH[3];
    XH[0] = (_Float16*)alloc((size_t)Bn * 128 * Nn * 2);
    XH[1] = (_Float16*)alloc((size_t)Bn * 320 * Nn * 2);
    XH[2] = (_Float16*)alloc((size_t)Bn * 512 * Nn * 2);
    _Float16* WH[3][3];           // effective f16 weights [branch][k/q/v]
    const int Cin[3] = {128, 320, 512};
    for (int i = 0; i < 3; ++i)
        for (int j = 0; j < 3; ++j)
            WH[i][j] = (_Float16*)alloc((size_t)Dn * Cin[i] * 2);
    _Float16* RW1 = (_Float16*)alloc((size_t)128 * Dn * 2);
    _Float16* RW2 = (_Float16*)alloc((size_t)320 * Dn * 2);
    float* BE[2][3];              // effective biases for branches 1,2
    for (int i = 0; i < 2; ++i)
        for (int j = 0; j < 3; ++j) BE[i][j] = (float*)alloc(Dn * 4);
    float* Pbuf = (float*)alloc(64);
    float* Gbuf = (float*)alloc(64);
    _Float16* KH   = (_Float16*)alloc((size_t)Bn * Dn * Nn * 2);
    _Float16* VH   = (_Float16*)alloc((size_t)Bn * Dn * Nn * 2);
    _Float16* QH   = (_Float16*)alloc((size_t)Bn * Dn * Nn * 2);
    float*    CTX  = (float*)   alloc((size_t)3 * Bn * CTXE * 4);
    _Float16* CTXT = (_Float16*)alloc((size_t)Bn * CTXE * 2);
    _Float16* AH   = (_Float16*)alloc((size_t)Bn * Dn * Nn * 2);
    (void)ws_size; (void)n_in; (void)in_sizes; (void)out_size;

    // output slices (floats)
    float* out1 = (float*)d_out;                         // [4,128,64,64]
    float* out2 = out1 + (size_t)Bn * 128 * Nn;          // [4,320,64,64]
    float* out3 = out2 + (size_t)Bn * 320 * Nn;          // [4,512,64,64]

    // ---- phase 1: convert inputs to f16
    f32_to_f16<<<4096, 256, 0, stream>>>(x1, XH[0], (size_t)Bn * 128 * Nn);
    f32_to_f16<<<4096, 256, 0, stream>>>(x2, XH[1], (size_t)Bn * 320 * Nn);
    f32_to_f16<<<4096, 256, 0, stream>>>(x3, XH[2], (size_t)Bn * 512 * Nn);

    // ---- phase 2: effective weights/biases
    const float* mapw[2] = {map1_w, map2_w};
    const float* mapb[2] = {map1_b, map2_b};
    for (int i = 0; i < 2; ++i)
        for (int j = 0; j < 3; ++j) {
            const int C = Cin[i];
            eff_weight<<<(512 * C + 255) / 256, 256, 0, stream>>>(kqv_w[i][j], mapw[i], WH[i][j], C);
            eff_bias<<<2, 256, 0, stream>>>(kqv_w[i][j], mapb[i], kqv_b[i][j], BE[i][j]);
        }
    for (int j = 0; j < 3; ++j)  // branch 3: plain convert
        f32_to_f16<<<1024, 256, 0, stream>>>(kqv_w[2][j], WH[2][j], (size_t)Dn * Dn);
    f32_to_f16<<<256, 256, 0, stream>>>(rev1_w, RW1, (size_t)128 * Dn);
    f32_to_f16<<<640, 256, 0, stream>>>(rev2_w, RW2, (size_t)320 * Dn);

    const float* biasK[3] = {BE[0][0], BE[1][0], kqv_b[2][0]};
    const float* biasQ[3] = {BE[0][1], BE[1][1], kqv_b[2][1]};
    const float* biasV[3] = {BE[0][2], BE[1][2], kqv_b[2][2]};

    const size_t sX[3] = {(size_t)128 * Nn, (size_t)320 * Nn, (size_t)512 * Nn};
    const size_t sDN   = (size_t)Dn * Nn;
    const dim3 gProj(Nn / 256, Dn / 64, Bn);   // 16 x 8 x 4
    const dim3 gCtx(Dn / 256, Dn / 64, Bn);    // 2 x 8 x 4

    // ---- phase 3: K, V, spatial softmax, ctx_i = K · Vᵀ
    for (int i = 0; i < 3; ++i) {
        gemm_wmma<false, true><<<gProj, 256, 0, stream>>>(
            WH[i][0], 0, Cin[i], XH[i], sX[i], Nn, KH, sDN, Nn, biasK[i], Cin[i]);
        gemm_wmma<false, true><<<gProj, 256, 0, stream>>>(
            WH[i][2], 0, Cin[i], XH[i], sX[i], Nn, VH, sDN, Nn, biasV[i], Cin[i]);
        softmax_row<<<Bn * Dn, 256, 0, stream>>>(KH);
        gemm_wmma<true, false><<<gCtx, 256, 0, stream>>>(
            KH, sDN, Nn, VH, sDN, Nn, CTX + (size_t)i * Bn * CTXE, CTXE, Dn,
            nullptr, Nn);
    }

    // ---- phase 4: SE gate + combined (transposed) context
    se_pool<<<3 * Bn, 256, 0, stream>>>(CTX, Pbuf);
    se_gate<<<1, 32, 0, stream>>>(Pbuf, se_fc1_w, se_fc1_b, se_fc2_w, se_fc2_b, Gbuf);
    se_combine<<<(unsigned)((Bn * CTXE) / 256), 256, 0, stream>>>(
        CTX, Gbuf, se_conv_w, se_conv_b, CTXT);

    // ---- phase 5: Q, channel softmax, attend, reverse convs
    for (int i = 0; i < 3; ++i) {
        gemm_wmma<false, true><<<gProj, 256, 0, stream>>>(
            WH[i][1], 0, Cin[i], XH[i], sX[i], Nn, QH, sDN, Nn, biasQ[i], Cin[i]);
        softmax_col<<<dim3(Nn / 256, Bn), 256, 0, stream>>>(QH);
        if (i == 2) {
            // a3 straight to output (f32)
            gemm_wmma<false, false><<<gProj, 256, 0, stream>>>(
                CTXT, CTXE, Dn, QH, sDN, Nn, out3, sDN, Nn, nullptr, Dn);
        } else {
            gemm_wmma<false, true><<<gProj, 256, 0, stream>>>(
                CTXT, CTXE, Dn, QH, sDN, Nn, AH, sDN, Nn, nullptr, Dn);
            const int M = (i == 0) ? 128 : 320;
            const _Float16* RW = (i == 0) ? RW1 : RW2;
            const float* rb = (i == 0) ? rev1_b : rev2_b;
            float* outp = (i == 0) ? out1 : out2;
            gemm_wmma<false, false><<<dim3(Nn / 256, M / 64, Bn), 256, 0, stream>>>(
                RW, 0, Dn, AH, sDN, Nn, outp, (size_t)M * Nn, Nn, rb, Dn);
        }
    }
}
